// GraphAttentionLayer_32976758899214
// MI455X (gfx1250) — compile-verified, matrix operand order checked by reading, not run on hardware
//
#include <hip/hip_runtime.h>
#include <hip/hip_bf16.h>

typedef __attribute__((ext_vector_type(16))) _Float16 v16h;
typedef __attribute__((ext_vector_type(8)))  _Float16 v8h;
typedef __attribute__((ext_vector_type(8)))  float    v8f;
typedef __attribute__((ext_vector_type(4)))  int      v4i_;

#define B_  16
#define N_  1024
#define F_  256
#define O_  256
#define H_  4
#define D_  64
#define NEG_SLOPE 0.2f
#define NEG_BIG  -3.0e38f

#if __has_builtin(__builtin_amdgcn_global_load_async_to_lds_b128) && \
    __has_builtin(__builtin_amdgcn_s_wait_asynccnt)
#define HAVE_ASYNC_LDS 1
#else
#define HAVE_ASYNC_LDS 0
#endif

__device__ __forceinline__ float lrelu(float x) { return x >= 0.f ? x : NEG_SLOPE * x; }

#if HAVE_ASYNC_LDS
// Async global->LDS 16B copy (ASYNCcnt-tracked, no VGPR data round-trip).
// Builtin signature (from the compiler diagnostic): (int4 as1*, int4 as3*, imm, imm).
__device__ __forceinline__ void async_cp16(const void* g, void* l)
{
  v4i_* gg = (v4i_*)g;   // reinterpret pointee in generic AS (drops const)
  v4i_* ll = (v4i_*)l;
  __builtin_amdgcn_global_load_async_to_lds_b128(
      (__attribute__((address_space(1))) v4i_*)gg,   // pure AS conversion
      (__attribute__((address_space(3))) v4i_*)ll,
      /*offset=*/0, /*cpol=*/0);
}
#endif

// ---------------------------------------------------------------------------
// Phase 0: one-shot f32 -> f16 conversion (h and W), streaming, 8 elem/thread.
// ---------------------------------------------------------------------------
__global__ void __launch_bounds__(256)
cvt_f16(const float* __restrict__ src, _Float16* __restrict__ dst, int n)
{
  const int i = (blockIdx.x * 256 + threadIdx.x) * 8;
  if (i + 8 <= n) {
    float4 x = *(const float4*)(src + i);
    float4 y = *(const float4*)(src + i + 4);
    v8h o;
    o[0] = (_Float16)x.x; o[1] = (_Float16)x.y; o[2] = (_Float16)x.z; o[3] = (_Float16)x.w;
    o[4] = (_Float16)y.x; o[5] = (_Float16)y.y; o[6] = (_Float16)y.z; o[7] = (_Float16)y.w;
    *(v8h*)(dst + i) = o;
  }
}

// ---------------------------------------------------------------------------
// Phase 1: Wh = h @ W^T via v_wmma_f32_16x16x32_f16 (f16 in, f32 accum).
//   grid.x = (B*N)/16 row tiles, block = 128 threads = 4 waves, wave -> head.
//   Epilogue: WhT[b][h][d][n] f16 (transposed for phase-3), s1, s2 (+mask).
// ---------------------------------------------------------------------------
__global__ void __launch_bounds__(128)
gat_gemm1(const _Float16* __restrict__ h16, const int* __restrict__ mask,
          const _Float16* __restrict__ W16, const float* __restrict__ a,
          _Float16* __restrict__ WhT, float* __restrict__ s1, float* __restrict__ s2)
{
  const int lane  = threadIdx.x & 31;
  const int head  = threadIdx.x >> 5;      // 0..3
  const int lmod  = lane & 15;
  const int lhalf = lane >> 4;
  const int koff  = lhalf * 8;             // per-half K split {0..7,16..23}/{8..15,24..31}

  const int row0 = blockIdx.x * 16;        // row in flattened (B*N, F)
  const int b    = row0 >> 10;
  const int n0   = row0 & (N_ - 1);
  const int bh   = b * H_ + head;

  v8f acc[4] = {};

  const _Float16* __restrict__ hrow = h16 + (size_t)(row0 + lmod) * F_;

#pragma unroll
  for (int kk = 0; kk < F_; kk += 32) {
    v16h afrag;
    {
      v8h lo = *(const v8h*)(hrow + kk + koff);
      v8h hi = *(const v8h*)(hrow + kk + 16 + koff);
#pragma unroll
      for (int t = 0; t < 8; ++t) { afrag[t] = lo[t]; afrag[8 + t] = hi[t]; }
    }
#pragma unroll
    for (int ct = 0; ct < 4; ++ct) {
      const _Float16* __restrict__ wrow = W16 + (size_t)(head * D_ + ct * 16 + lmod) * F_;
      v16h bfrag;
      {
        v8h lo = *(const v8h*)(wrow + kk + koff);
        v8h hi = *(const v8h*)(wrow + kk + 16 + koff);
#pragma unroll
        for (int t = 0; t < 8; ++t) { bfrag[t] = lo[t]; bfrag[8 + t] = hi[t]; }
      }
      acc[ct] = __builtin_amdgcn_wmma_f32_16x16x32_f16(false, afrag, false, bfrag,
                                                       (short)0, acc[ct], false, false);
    }
  }

  // WhT (f16, [b][h][d][n]): lane holds col d, rows 8*lhalf+r -> one 16B store per d.
#pragma unroll
  for (int ct = 0; ct < 4; ++ct) {
    v8h pk;
#pragma unroll
    for (int r = 0; r < 8; ++r) pk[r] = (_Float16)acc[ct][r];
    const int d = ct * 16 + lmod;
    *(v8h*)(WhT + ((size_t)bh * D_ + d) * N_ + (n0 + 8 * lhalf)) = pk;
  }

  // s1 = Wh.a1, s2 = Wh.a2 from f32 accumulators.
  float p1[8], p2[8];
#pragma unroll
  for (int r = 0; r < 8; ++r) { p1[r] = 0.f; p2[r] = 0.f; }
#pragma unroll
  for (int ct = 0; ct < 4; ++ct) {
    const float a1v = a[ct * 16 + lmod];
    const float a2v = a[D_ + ct * 16 + lmod];
#pragma unroll
    for (int r = 0; r < 8; ++r) { p1[r] += acc[ct][r] * a1v; p2[r] += acc[ct][r] * a2v; }
  }
#pragma unroll
  for (int m = 1; m < 16; m <<= 1) {
#pragma unroll
    for (int r = 0; r < 8; ++r) {
      p1[r] += __shfl_xor(p1[r], m, 32);
      p2[r] += __shfl_xor(p2[r], m, 32);
    }
  }
  if (lmod == 0) {
#pragma unroll
    for (int r = 0; r < 8; ++r) {
      const int n = n0 + 8 * lhalf + r;
      s1[(size_t)bh * N_ + n] = p1[r];
      s2[(size_t)bh * N_ + n] = mask[b * N_ + n] ? p2[r] : NEG_BIG;
    }
  }
}

// ---------------------------------------------------------------------------
// Phase 2: m2[bh] = max_j s2[bh][j]  (mask already folded into s2)
// ---------------------------------------------------------------------------
__global__ void __launch_bounds__(256)
gat_s2max(const float* __restrict__ s2, float* __restrict__ m2)
{
  const int bh = blockIdx.x;
  const int tid = threadIdx.x;
  float v = NEG_BIG;
  for (int j = tid; j < N_; j += 256) v = fmaxf(v, s2[(size_t)bh * N_ + j]);
#pragma unroll
  for (int m = 16; m >= 1; m >>= 1) v = fmaxf(v, __shfl_xor(v, m, 32));
  __shared__ float red[8];
  if ((tid & 31) == 0) red[tid >> 5] = v;
  __syncthreads();
  if (tid == 0) {
    float r = red[0];
#pragma unroll
    for (int i = 1; i < 8; ++i) r = fmaxf(r, red[i]);
    m2[bh] = r;
  }
}

// ---------------------------------------------------------------------------
// Phase 3: fused softmax + (attn @ Wh). Block = 8 waves, one (b,h) slice:
//   stage full WhT[bh] (64 x 1024 f16 = 128 KB, +16B row pad for conflict-free
//   fragment reads) + s2 row (4 KB) into LDS — via the gfx1250 async
//   global->LDS engine (ASYNCcnt) when available.
//   Each wave owns one 16-row i-tile; row max known a priori -> single pass.
// ---------------------------------------------------------------------------
__global__ void __launch_bounds__(256)
gat_attn(const _Float16* __restrict__ WhT, const float* __restrict__ s1,
         const float* __restrict__ s2, const float* __restrict__ m2,
         const int* __restrict__ mask, float* __restrict__ out)
{
  const int tid   = threadIdx.x;
  const int wid   = tid >> 5;               // 0..7
  const int lane  = tid & 31;
  const int lmod  = lane & 15;
  const int lhalf = lane >> 4;
  const int koff  = lhalf * 8;

  const int i0   = (blockIdx.x * 8 + wid) * 16;
  const int head = blockIdx.y;
  const int b    = blockIdx.z;
  const int bh   = b * H_ + head;

  __shared__ _Float16 whl[D_][N_ + 8];      // 132 KB, padded row pitch (2064 B)
  __shared__ float    s2l[N_];              // 4 KB

  const _Float16* __restrict__ whbase = WhT + (size_t)bh * D_ * N_;
  for (int j = tid; j < N_; j += 256) s2l[j] = s2[(size_t)bh * N_ + j];

#if HAVE_ASYNC_LDS
#pragma unroll 4
  for (int c = tid; c < (D_ * N_) / 8; c += 256) {
    const int d  = c >> 7;                  // 128 8-half chunks per row
    const int jc = (c & 127) * 8;
    async_cp16(whbase + (size_t)d * N_ + jc, &whl[d][jc]);
  }
  __builtin_amdgcn_s_wait_asynccnt(0);
#else
  for (int c = tid; c < (D_ * N_) / 8; c += 256) {
    const int d  = c >> 7;
    const int jc = (c & 127) * 8;
    *(v8h*)&whl[d][jc] = *(const v8h*)(whbase + (size_t)d * N_ + jc);
  }
#endif
  __syncthreads();

  const float mrow = m2[bh];
  const bool  dead = (mrow < -1.0e37f);
  const float s1v  = s1[(size_t)bh * N_ + i0 + lmod];
  const float mi   = lrelu(s1v + mrow);     // exact row max (lrelu monotone)

  v8f acc[4] = {};
  float lsum = 0.f;

  for (int j0 = 0; j0 < N_; j0 += 32) {
    v16h afrag;
#pragma unroll
    for (int t = 0; t < 16; ++t) {
      const int j = j0 + ((t < 8) ? (koff + t) : (16 + koff + (t - 8)));
      const float s2v = s2l[j];             // LDS broadcast within half-wave
      float p = 0.f;
      if (!dead && s2v > -1.0e37f) p = __expf(lrelu(s1v + s2v) - mi);  // <= 1
      const _Float16 ph = (_Float16)p;
      lsum += (float)ph;
      afrag[t] = ph;
    }
#pragma unroll
    for (int ct = 0; ct < 4; ++ct) {
      const _Float16* wd = &whl[ct * 16 + lmod][j0];
      v8h lo = *(const v8h*)(wd + koff);        // ds_load_b128, conflict-free
      v8h hi = *(const v8h*)(wd + 16 + koff);
      v16h bfrag;
#pragma unroll
      for (int t = 0; t < 8; ++t) { bfrag[t] = lo[t]; bfrag[8 + t] = hi[t]; }
      acc[ct] = __builtin_amdgcn_wmma_f32_16x16x32_f16(false, afrag, false, bfrag,
                                                       (short)0, acc[ct], false, false);
    }
  }

  // Denominator per row: lanes l and l+16 cover complementary K halves.
  lsum += __shfl_xor(lsum, 16, 32);
  float inv = (lsum > 0.f) ? (1.f / lsum) : 0.f;
  if (!mask[b * N_ + i0 + lmod]) inv = 0.f;     // fold output row mask

  float invr[8];
#pragma unroll
  for (int r = 0; r < 8; ++r) invr[r] = __shfl(inv, 8 * lhalf + r, 32);

#pragma unroll
  for (int ct = 0; ct < 4; ++ct) {
    const int d = head * D_ + ct * 16 + lmod;
#pragma unroll
    for (int r = 0; r < 8; ++r) {
      const int irow = i0 + 8 * lhalf + r;
      out[((size_t)b * N_ + irow) * O_ + d] = acc[ct][r] * invr[r];
    }
  }
}

// ---------------------------------------------------------------------------
extern "C" void kernel_launch(void* const* d_in, const int* in_sizes, int n_in,
                              void* d_out, int out_size, void* d_ws, size_t ws_size,
                              hipStream_t stream) {
  (void)in_sizes; (void)n_in; (void)out_size; (void)ws_size;
  const float* h    = (const float*)d_in[0];
  const int*   mask = (const int*)d_in[1];
  const float* W    = (const float*)d_in[2];
  const float* a    = (const float*)d_in[3];
  float* out = (float*)d_out;

  char* ws = (char*)d_ws;
  _Float16* WhT = (_Float16*)ws;                                   // 8 MB
  _Float16* h16 = (_Float16*)(ws + (size_t)8 * 1024 * 1024);       // 8 MB
  _Float16* W16 = (_Float16*)(ws + (size_t)16 * 1024 * 1024);      // 128 KB
  float* s1 = (float*)(ws + (size_t)16 * 1024 * 1024 + 128 * 1024);
  float* s2 = s1 + (size_t)B_ * H_ * N_;
  float* m2 = s2 + (size_t)B_ * H_ * N_;

  const int nh = B_ * N_ * F_;   // 4,194,304
  const int nw = O_ * F_;        // 65,536
  cvt_f16<<<dim3(nh / (256 * 8)), dim3(256), 0, stream>>>(h, h16, nh);
  cvt_f16<<<dim3(nw / (256 * 8)), dim3(256), 0, stream>>>(W, W16, nw);
  gat_gemm1<<<dim3((B_ * N_) / 16), dim3(128), 0, stream>>>(h16, mask, W16, a, WhT, s1, s2);
  gat_s2max<<<dim3(B_ * H_), dim3(256), 0, stream>>>(s2, m2);
  gat_attn<<<dim3(N_ / (16 * 8), H_, B_), dim3(256), 0, stream>>>(WhT, s1, s2, m2, mask, out);
}
